// UniformQuantizedLinear_64330020159907
// MI455X (gfx1250) — matrix operand order repaired and checked by reading, not compiled, not run on hardware
//
#include <hip/hip_runtime.h>

// Problem constants (match the reference)
#define KOUT   4096   // output features (reference K)
#define NRED   4096   // reduction dim (reference N)
#define MROWS  8192   // BATCH * SEQ
#define GROUP  128
#define NGRP   32

// GEMM tiling
#define BM 256                    // M rows per block
#define BC 128                    // output cols per block
#define KS 32                     // reduction step (WMMA K)
#define LROW 80                   // LDS bytes per row: 64 data + 16 pad (bank-conflict free)
#define XTILE_BYTES (BM * LROW)   // 20480
#define WTILE_BYTES (BC * LROW)   // 10240
#define STAGE_BYTES (XTILE_BYTES + WTILE_BYTES)   // 30720

typedef __attribute__((ext_vector_type(16))) __bf16 v16bf;
typedef __attribute__((ext_vector_type(8)))  float  v8f;

static __device__ __forceinline__ unsigned short f2bf16(float f) {
  union { float f; unsigned int u; } c; c.f = f;
  unsigned int r = 0x7FFFu + ((c.u >> 16) & 1u);
  return (unsigned short)((c.u + r) >> 16);
}

// Async 16B copy global -> LDS, tracked by ASYNCcnt (per-lane, GV addressing).
static __device__ __forceinline__ void async_load_b128(unsigned lds_off, const void* gaddr) {
  asm volatile("global_load_async_to_lds_b128 %0, %1, off"
               :: "v"(lds_off), "v"(gaddr)
               : "memory");
}

static __device__ __forceinline__ void wait_asynccnt0() {
#if __has_builtin(__builtin_amdgcn_s_wait_asynccnt)
  __builtin_amdgcn_s_wait_asynccnt(0);
#else
  asm volatile("s_wait_asynccnt 0x0" ::: "memory");
#endif
}

// ---------------------------------------------------------------------------
// Kernel 1: dequantize W_q (int32, values 0..15) -> bf16 W_deq [K, N]
// W_deq[k,n] = (q - zeros[k,g]) * scales[k,g] * mu2[k] * mu1[n]
// ---------------------------------------------------------------------------
__global__ __launch_bounds__(256) void dequant_w_kernel(
    const int* __restrict__ Wq, const float* __restrict__ scales,
    const float* __restrict__ zeros, const float* __restrict__ mu1,
    const float* __restrict__ mu2, unsigned short* __restrict__ Wbf) {
  const unsigned gid  = blockIdx.x * 256u + threadIdx.x;
  const size_t   base = (size_t)gid * 8u;
  const unsigned k  = (unsigned)(base / NRED);
  const unsigned n0 = (unsigned)(base % NRED);
  const unsigned g  = n0 / GROUP;

  const float z  = zeros[(size_t)k * NGRP + g];
  const float sm = scales[(size_t)k * NGRP + g] * mu2[k];

  const int4* wq = (const int4*)(Wq + base);
  int4 q0 = wq[0];
  int4 q1 = wq[1];
  const float4* m1 = (const float4*)(mu1 + n0);
  float4 u0 = m1[0];
  float4 u1 = m1[1];

  float r0 = ((float)q0.x - z) * sm * u0.x;
  float r1 = ((float)q0.y - z) * sm * u0.y;
  float r2 = ((float)q0.z - z) * sm * u0.z;
  float r3 = ((float)q0.w - z) * sm * u0.w;
  float r4 = ((float)q1.x - z) * sm * u1.x;
  float r5 = ((float)q1.y - z) * sm * u1.y;
  float r6 = ((float)q1.z - z) * sm * u1.z;
  float r7 = ((float)q1.w - z) * sm * u1.w;

  uint4 packed;
  packed.x = (unsigned)f2bf16(r0) | ((unsigned)f2bf16(r1) << 16);
  packed.y = (unsigned)f2bf16(r2) | ((unsigned)f2bf16(r3) << 16);
  packed.z = (unsigned)f2bf16(r4) | ((unsigned)f2bf16(r5) << 16);
  packed.w = (unsigned)f2bf16(r6) | ((unsigned)f2bf16(r7) << 16);
  *(uint4*)(Wbf + base) = packed;
}

// ---------------------------------------------------------------------------
// Kernel 2: convert x f32 -> bf16, 8 elements per thread
// ---------------------------------------------------------------------------
__global__ __launch_bounds__(256) void convert_x_kernel(
    const float* __restrict__ x, unsigned short* __restrict__ Xbf) {
  const unsigned gid  = blockIdx.x * 256u + threadIdx.x;
  const size_t   base = (size_t)gid * 8u;
  const float4* xp = (const float4*)(x + base);
  float4 a = xp[0];
  float4 b = xp[1];
  uint4 packed;
  packed.x = (unsigned)f2bf16(a.x) | ((unsigned)f2bf16(a.y) << 16);
  packed.y = (unsigned)f2bf16(a.z) | ((unsigned)f2bf16(a.w) << 16);
  packed.z = (unsigned)f2bf16(b.x) | ((unsigned)f2bf16(b.y) << 16);
  packed.w = (unsigned)f2bf16(b.z) | ((unsigned)f2bf16(b.w) << 16);
  *(uint4*)(Xbf + base) = packed;
}

// ---------------------------------------------------------------------------
// Kernel 3: bf16 WMMA GEMM, LDS double-buffered with async-to-LDS copies.
//   out[m, c] = sum_n Xbf[m, n] * Wbf[c, n] + bias[c]
// Block: 256 threads = 8 waves (4 x M, 2 x C). Block tile 256x128.
// Wave tile 64x64 = 4x4 WMMA 16x16 tiles; A fragments register-resident.
// LDS rows padded to 80B -> ds_load_b128 start banks 20*row mod 64,
// distinct and 4 apart for 16 rows => conflict-free.
// ---------------------------------------------------------------------------
__global__ __launch_bounds__(256) void wmma_gemm_lds_kernel(
    const unsigned short* __restrict__ Xbf,   // [MROWS, NRED]
    const unsigned short* __restrict__ Wbf,   // [KOUT,  NRED]
    const float* __restrict__ bias,           // [KOUT]
    float* __restrict__ out) {                // [MROWS, KOUT]
  __shared__ __align__(16) unsigned char lds[2 * STAGE_BYTES];

  const unsigned tid   = threadIdx.x;
  const unsigned lane  = tid & 31u;
  const unsigned wave  = tid >> 5u;
  const unsigned wm    = wave & 3u;            // 0..3 -> M
  const unsigned wc    = wave >> 2u;           // 0..1 -> C
  const unsigned mBase = blockIdx.y * BM;
  const unsigned cBase = blockIdx.x * BC;

  const unsigned lhalf = lane & 15u;
  const bool     hi    = (lane >= 16u);

  // copy-task constants: each thread moves 16B chunks
  const unsigned crow   = tid >> 2;            // 0..63
  const unsigned cchunk = tid & 3u;            // 0..3
  const unsigned ldsBase = (unsigned)(size_t)(&lds[0]);   // LDS byte offset of array

  union Frag { v16bf v; uint4 u[2]; };
  v8f acc[4][4] = {};

  // ---- stage copy: X tile (256 rows x 64B) + W tile (128 rows x 64B) ----
  auto copy_stage = [&](unsigned stage, unsigned kk) {
    const unsigned sBase = ldsBase + stage * STAGE_BYTES;
#pragma unroll
    for (unsigned i = 0; i < 4; ++i) {                    // X: 1024 chunks / 256 thr
      const unsigned row = crow + i * 64u;
      const void* g = (const void*)(Xbf + (size_t)(mBase + row) * NRED + kk + cchunk * 8u);
      async_load_b128(sBase + row * LROW + cchunk * 16u, g);
    }
#pragma unroll
    for (unsigned i = 0; i < 2; ++i) {                    // W: 512 chunks / 256 thr
      const unsigned row = crow + i * 64u;
      const void* g = (const void*)(Wbf + (size_t)(cBase + row) * NRED + kk + cchunk * 8u);
      async_load_b128(sBase + XTILE_BYTES + row * LROW + cchunk * 16u, g);
    }
  };

  // prologue
  copy_stage(0u, 0u);
  wait_asynccnt0();
  __syncthreads();

  unsigned cur = 0u;
  for (unsigned kk = 0; kk < NRED; kk += KS) {
    if (kk + KS < NRED) copy_stage(cur ^ 1u, kk + KS);    // prefetch next stage

    const unsigned stOff = cur * STAGE_BYTES;

    // A fragments (register resident, reused across 4 B tiles)
    // lanes 0-15: K bytes [0,16) and [32,48); lanes 16-31: [16,32) and [48,64)
    Frag a[4];
    const unsigned aSel = hi ? 16u : 0u;
#pragma unroll
    for (unsigned it = 0; it < 4; ++it) {
      const unsigned row = wm * 64u + it * 16u + lhalf;
      const unsigned off = stOff + row * LROW + aSel;
      a[it].u[0] = *(const uint4*)(lds + off);
      a[it].u[1] = *(const uint4*)(lds + off + 32u);
    }

    // B fragments streamed: lanes 0-15 K bytes [0,16) and [32,48); lanes 16-31: [16,32) and [48,64)
    const unsigned bSel = hi ? 16u : 0u;
#pragma unroll
    for (unsigned jt = 0; jt < 4; ++jt) {
      const unsigned row = wc * 64u + jt * 16u + lhalf;
      const unsigned off = stOff + XTILE_BYTES + row * LROW + bSel;
      Frag b;
      b.u[0] = *(const uint4*)(lds + off);
      b.u[1] = *(const uint4*)(lds + off + 32u);
#pragma unroll
      for (unsigned it = 0; it < 4; ++it) {
        acc[it][jt] = __builtin_amdgcn_wmma_f32_16x16x32_bf16(
            false, a[it].v, false, b.v, (short)0, acc[it][jt], false, false);
      }
    }

    wait_asynccnt0();     // next-stage copies have landed
    __syncthreads();      // all waves done reading cur & copies visible
    cur ^= 1u;
  }

  // ---- epilogue: bias add + store ----
  // C/D layout: VGPR r -> M = r (lanes 0-15) or r+8 (lanes 16-31); N = lane%16
  const unsigned rsel = hi ? 8u : 0u;
#pragma unroll
  for (unsigned jt = 0; jt < 4; ++jt) {
    const unsigned col = cBase + wc * 64u + jt * 16u + lhalf;
    const float bb = bias[col];
#pragma unroll
    for (unsigned it = 0; it < 4; ++it) {
      const unsigned rbase = mBase + wm * 64u + it * 16u + rsel;
#pragma unroll
      for (int r = 0; r < 8; ++r) {
        out[(size_t)(rbase + (unsigned)r) * KOUT + col] = acc[it][jt][r] + bb;
      }
    }
  }
}

// ---------------------------------------------------------------------------
extern "C" void kernel_launch(void* const* d_in, const int* in_sizes, int n_in,
                              void* d_out, int out_size, void* d_ws, size_t ws_size,
                              hipStream_t stream) {
  const float* x      = (const float*)d_in[0];
  const int*   Wq     = (const int*)  d_in[1];
  const float* scales = (const float*)d_in[2];
  const float* zeros  = (const float*)d_in[3];
  const float* mu1    = (const float*)d_in[4];
  const float* mu2    = (const float*)d_in[5];
  const float* bias   = (const float*)d_in[6];
  float* out = (float*)d_out;

  unsigned short* Wbf = (unsigned short*)d_ws;                     // 32 MiB
  unsigned short* Xbf = Wbf + (size_t)KOUT * NRED;                 // 64 MiB

  {
    const unsigned blocks = (KOUT * (unsigned)NRED / 8u) / 256u;   // 8192
    dequant_w_kernel<<<blocks, 256, 0, stream>>>(Wq, scales, zeros, mu1, mu2, Wbf);
  }
  {
    const unsigned blocks = ((unsigned)MROWS * (unsigned)NRED / 8u) / 256u; // 16384
    convert_x_kernel<<<blocks, 256, 0, stream>>>(x, Xbf);
  }
  {
    dim3 grid(KOUT / BC, MROWS / BM);                              // 32 x 32
    wmma_gemm_lds_kernel<<<grid, 256, 0, stream>>>(Xbf, Wbf, bias, out);
  }
}
